// MultiheadAttentionWithVdim_54786602827873
// MI455X (gfx1250) — matrix operand order repaired and checked by reading, not compiled, not run on hardware
//
#include <hip/hip_runtime.h>
#include <hip/hip_bf16.h>

// MultiheadAttentionWithVdim for MI455X (gfx1250, wave32, WMMA bf16, fp32 acc).
// [Q/K/V projections: WMMA GEMM, bf16 head-split out] ->
// [flash attention: 4-wave blocks, shared K/V staging (async-to-LDS), online
//  softmax, WMMA for QK^T and PV] -> [Wo projection: WMMA GEMM, fp32 out].

typedef __attribute__((ext_vector_type(16))) __bf16 v16bf;
typedef __attribute__((ext_vector_type(8)))  float  v8f;
typedef __attribute__((ext_vector_type(4)))  float  f32x4;
typedef __attribute__((ext_vector_type(4)))  __bf16 bf16x4;
typedef __attribute__((ext_vector_type(4)))  int    i32x4;

#define B_DIM  8
#define LQ     900
#define LK     4096
#define E_DIM  512
#define V_DIM  256
#define H_DIM  8
#define HD     64
#define VD     32
#define SCALE  0.125f   // 64^-0.5

#if defined(__has_builtin)
#if __has_builtin(__builtin_amdgcn_global_load_async_to_lds_b128) && \
    __has_builtin(__builtin_amdgcn_s_wait_asynccnt)
#define USE_ASYNC_LDS 1
#endif
#endif

#ifdef USE_ASYNC_LDS
// Builtin signature (from hipcc diagnostic): (int4 AS1*, int4 AS3*, imm, imm).
__device__ __forceinline__ void async_cp_b128(const void* g, void* l) {
  __builtin_amdgcn_global_load_async_to_lds_b128(
      (__attribute__((address_space(1))) i32x4*)(i32x4*)(void*)g,
      (__attribute__((address_space(3))) i32x4*)(i32x4*)l,
      /*offset=*/0, /*cpol=*/0);
}
#endif

// A-fragment (16x32 bf16) K index for element e (0..15), lane-half hf (0/1).
// ISA 7.12.2: half 0 holds K {0..7,16..23}; half 1 holds {8..15,24..31}.
__device__ __forceinline__ int amapK(int e, int hf) {
  return e + ((e & 8) ? 8 : 0) + 8 * hf;
}

// ---------------------------------------------------------------------------
// C[M,N] = A[M,K] @ W[N,K]^T + bias.  One wave computes a 16x64 tile.
// outH != nullptr: bf16 head-split layout [B][H=N/D][L][D]. Else fp32 outF.
// ---------------------------------------------------------------------------
__global__ __launch_bounds__(32)
void gemm_wmma_bf16(const float* __restrict__ A, const float* __restrict__ W,
                    const float* __restrict__ bias, int M, int N, int K,
                    float* __restrict__ outF, __bf16* __restrict__ outH,
                    int L, int D) {
  __shared__ __bf16 lds_a[16 * 32];
  __shared__ __bf16 lds_w[64 * 32];
  const int lane = threadIdx.x;
  const int mrow = lane & 15;
  const int hf   = lane >> 4;
  const int m0 = blockIdx.x * 16;
  const int n0 = blockIdx.y * 64;

  v8f acc[4];
  for (int j = 0; j < 4; ++j)
    for (int r = 0; r < 8; ++r) acc[j][r] = 0.0f;

  for (int k0 = 0; k0 < K; k0 += 32) {
    for (int t = 0; t < 4; ++t) {                // stage A 16x32 (float4 loads)
      int g = t * 32 + lane;
      int r = g >> 3, c4 = (g & 7) * 4;
      int m = m0 + r;
      f32x4 v = {0.0f, 0.0f, 0.0f, 0.0f};
      if (m < M) v = *(const f32x4*)(A + (size_t)m * K + k0 + c4);
      bf16x4 o;
#pragma unroll
      for (int i = 0; i < 4; ++i) o[i] = (__bf16)v[i];
      *(bf16x4*)(lds_a + g * 4) = o;
    }
    for (int t = 0; t < 16; ++t) {               // stage W 64x32
      int g = t * 32 + lane;
      int n = g >> 3, c4 = (g & 7) * 4;
      f32x4 v = *(const f32x4*)(W + (size_t)(n0 + n) * K + k0 + c4);
      bf16x4 o;
#pragma unroll
      for (int i = 0; i < 4; ++i) o[i] = (__bf16)v[i];
      *(bf16x4*)(lds_w + g * 4) = o;
    }
    __syncthreads();

    v16bf a;
#pragma unroll
    for (int e = 0; e < 16; ++e)
      a[e] = lds_a[mrow * 32 + (e + 16 * hf)];
#pragma unroll
    for (int j = 0; j < 4; ++j) {
      v16bf b;
#pragma unroll
      for (int e = 0; e < 16; ++e)               // B[k][n] = W[n][k]
        b[e] = lds_w[(j * 16 + mrow) * 32 + (e + 16 * hf)];
      acc[j] = __builtin_amdgcn_wmma_f32_16x16x32_bf16(
          false, a, false, b, (short)0, acc[j], false, false);
    }
    __syncthreads();
  }

  for (int j = 0; j < 4; ++j)
    for (int r = 0; r < 8; ++r) {
      int m = m0 + r + 8 * hf;                   // C layout: row = r + 8*half
      int n = n0 + j * 16 + mrow;                // col = lane&15
      if (m < M) {
        float v = acc[j][r] + bias[n];
        if (outH) {
          int bb = m / L, l = m % L;
          int hh = n / D, d = n % D;
          int Hh = N / D;
          outH[(((size_t)bb * Hh + hh) * L + l) * D + d] = (__bf16)v;
        } else {
          outF[(size_t)m * N + n] = v;
        }
      }
    }
}

// ---------------------------------------------------------------------------
// Flash attention. Block = 128 threads (4 waves); each wave owns a 16-row Q
// tile of a shared 64-row block so the K/V chunk staging is amortized 4x.
// K/V chunks go through GLOBAL_LOAD_ASYNC_TO_LDS when available.
// ---------------------------------------------------------------------------
__global__ __launch_bounds__(128)
void flash_attn(const __bf16* __restrict__ Qh,   // [B,H,Lq,HD] bf16
                const __bf16* __restrict__ Kh,   // [B,H,Lk,HD] bf16
                const __bf16* __restrict__ Vh,   // [B,H,Lk,VD] bf16
                const float* __restrict__ amask, // [Lq,Lk] fp32
                const unsigned char* __restrict__ kpm, // [B,Lk] bool
                float* __restrict__ O) {         // [B,Lq,H*VD] fp32
  __shared__ __bf16 q_lds[64 * 64];              // 8 KB
  __shared__ __bf16 k_lds[32 * 64];              // 4 KB
  __shared__ __bf16 v_lds[32 * 32];              // 2 KB
  __shared__ __bf16 p_lds[4][16 * 32];           // 4 KB (per-wave P tiles)
  const int tid  = threadIdx.x;
  const int wave = tid >> 5, lane = tid & 31;
  const int mrow = lane & 15, hf = lane >> 4;
  const int bh = blockIdx.y;
  const int b = bh / H_DIM, h = bh % H_DIM;
  const int qblk = blockIdx.x * 64;
  const int q0 = qblk + wave * 16;
  const size_t qbase = (size_t)bh * LQ * HD;
  const size_t kbase = (size_t)bh * LK * HD;
  const size_t vbase = (size_t)bh * LK * VD;

  for (int t = 0; t < 8; ++t) {                  // stage Q 64x64 cooperatively
    int g = t * 128 + tid;                       // 1024 groups of 4 halves
    int r = g >> 4, c4 = (g & 15) * 4;
    int q = qblk + r;
    bf16x4 v4;
    if (q < LQ) {
      v4 = *(const bf16x4*)(Qh + qbase + (size_t)q * HD + c4);
    } else {
#pragma unroll
      for (int i = 0; i < 4; ++i) v4[i] = (__bf16)0.0f;
    }
    *(bf16x4*)(q_lds + g * 4) = v4;
  }
  __syncthreads();

  v16bf aq0, aq1;                                // per-wave Q fragments
#pragma unroll
  for (int e = 0; e < 16; ++e) {
    aq0[e] = q_lds[(wave * 16 + mrow) * 64 + (e + 16 * hf)];       // dims 0..31
    aq1[e] = q_lds[(wave * 16 + mrow) * 64 + 32 + (e + 16 * hf)];  // dims 32..63
  }

  float m_r[8], l_r[8];
  v8f o0, o1;
  for (int r = 0; r < 8; ++r) { m_r[r] = -1e30f; l_r[r] = 0.0f; o0[r] = 0.0f; o1[r] = 0.0f; }

  for (int k0 = 0; k0 < LK; k0 += 32) {
#ifdef USE_ASYNC_LDS
    // K chunk 32x64 halves: 256 b128 transfers; V chunk 32x32: 128 transfers.
    for (int t = 0; t < 2; ++t) {
      int g = t * 128 + tid;
      int r = g >> 3, c8 = (g & 7) * 8;
      async_cp_b128(Kh + kbase + (size_t)(k0 + r) * HD + c8, k_lds + g * 8);
    }
    {
      int r = tid >> 2, c8 = (tid & 3) * 8;
      async_cp_b128(Vh + vbase + (size_t)(k0 + r) * VD + c8, v_lds + tid * 8);
    }
    __builtin_amdgcn_s_wait_asynccnt(0);
#else
    for (int t = 0; t < 2; ++t) {
      int g = t * 128 + tid;
      int r = g >> 3, c8 = (g & 7) * 8;
      *(bf16x4*)(k_lds + g * 8)     = *(const bf16x4*)(Kh + kbase + (size_t)(k0 + r) * HD + c8);
      *(bf16x4*)(k_lds + g * 8 + 4) = *(const bf16x4*)(Kh + kbase + (size_t)(k0 + r) * HD + c8 + 4);
    }
    {
      int r = tid >> 2, c8 = (tid & 3) * 8;
      *(bf16x4*)(v_lds + tid * 8)     = *(const bf16x4*)(Vh + vbase + (size_t)(k0 + r) * VD + c8);
      *(bf16x4*)(v_lds + tid * 8 + 4) = *(const bf16x4*)(Vh + vbase + (size_t)(k0 + r) * VD + c8 + 4);
    }
#endif
    __syncthreads();

    v8f s[2];
#pragma unroll
    for (int nh = 0; nh < 2; ++nh) {             // S tile 16x32 = two 16x16
      v8f c; for (int r = 0; r < 8; ++r) c[r] = 0.0f;
      v16bf bf;
#pragma unroll
      for (int e = 0; e < 16; ++e)               // B[k][n] = K[n][k]  (K^T)
        bf[e] = k_lds[(nh * 16 + mrow) * 64 + (e + 16 * hf)];
      c = __builtin_amdgcn_wmma_f32_16x16x32_bf16(false, aq0, false, bf, (short)0, c, false, false);
#pragma unroll
      for (int e = 0; e < 16; ++e)
        bf[e] = k_lds[(nh * 16 + mrow) * 64 + 32 + (e + 16 * hf)];
      c = __builtin_amdgcn_wmma_f32_16x16x32_bf16(false, aq1, false, bf, (short)0, c, false, false);
      s[nh] = c;
    }

#pragma unroll
    for (int nh = 0; nh < 2; ++nh) {             // scale + attn_mask + padding
      int col = k0 + nh * 16 + mrow;
      bool pad = kpm[(size_t)b * LK + col] != 0;
#pragma unroll
      for (int r = 0; r < 8; ++r) {
        int q = q0 + r + 8 * hf;
        float sv = s[nh][r] * SCALE + ((q < LQ) ? amask[(size_t)q * LK + col] : 0.0f);
        if (pad) sv = -1e30f;
        s[nh][r] = sv;
      }
    }

#pragma unroll
    for (int r = 0; r < 8; ++r) {                // online softmax per row
      float vmax = fmaxf(s[0][r], s[1][r]);
      for (int off = 1; off < 16; off <<= 1)
        vmax = fmaxf(vmax, __shfl_xor(vmax, off, 16));
      float mnew  = fmaxf(m_r[r], vmax);
      float alpha = __expf(m_r[r] - mnew);
      float p0 = (s[0][r] < -1e29f) ? 0.0f : __expf(s[0][r] - mnew);
      float p1 = (s[1][r] < -1e29f) ? 0.0f : __expf(s[1][r] - mnew);
      float rs = p0 + p1;
      for (int off = 1; off < 16; off <<= 1)
        rs += __shfl_xor(rs, off, 16);
      l_r[r] = l_r[r] * alpha + rs;
      m_r[r] = mnew;
      o0[r] *= alpha; o1[r] *= alpha;
      p_lds[wave][(r + 8 * hf) * 32 + mrow]      = (__bf16)p0;  // C-layout -> LDS
      p_lds[wave][(r + 8 * hf) * 32 + 16 + mrow] = (__bf16)p1;
    }
    // p_lds is per-wave and LDS ops are in-order within a wave: no barrier.

    v16bf ap;                                    // re-gather P as A-fragment
#pragma unroll
    for (int e = 0; e < 16; ++e)
      ap[e] = p_lds[wave][mrow * 32 + (e + 16 * hf)];
    {
      v16bf bv;
#pragma unroll
      for (int e = 0; e < 16; ++e)               // B[k][n] = V[k][n]
        bv[e] = v_lds[(e + 16 * hf) * 32 + mrow];
      o0 = __builtin_amdgcn_wmma_f32_16x16x32_bf16(false, ap, false, bv, (short)0, o0, false, false);
#pragma unroll
      for (int e = 0; e < 16; ++e)
        bv[e] = v_lds[(e + 16 * hf) * 32 + 16 + mrow];
      o1 = __builtin_amdgcn_wmma_f32_16x16x32_bf16(false, ap, false, bv, (short)0, o1, false, false);
    }
    __syncthreads();                             // all waves done with k/v_lds
  }

#pragma unroll
  for (int r = 0; r < 8; ++r) {
    int q = q0 + r + 8 * hf;
    if (q < LQ) {
      float inv = 1.0f / l_r[r];
      size_t rowoff = ((size_t)b * LQ + q) * (H_DIM * VD) + (size_t)h * VD;
      O[rowoff + mrow]      = o0[r] * inv;
      O[rowoff + 16 + mrow] = o1[r] * inv;
    }
  }
}

// ---------------------------------------------------------------------------
extern "C" void kernel_launch(void* const* d_in, const int* in_sizes, int n_in,
                              void* d_out, int out_size, void* d_ws, size_t ws_size,
                              hipStream_t stream) {
  (void)in_sizes; (void)n_in; (void)out_size; (void)ws_size;
  const float* query = (const float*)d_in[0];
  const float* key   = (const float*)d_in[1];
  const float* value = (const float*)d_in[2];
  const unsigned char* kpm = (const unsigned char*)d_in[3];  // numpy bool = 1B
  const float* amask = (const float*)d_in[4];
  const float* Wq = (const float*)d_in[5];  const float* bq = (const float*)d_in[6];
  const float* Wk = (const float*)d_in[7];  const float* bk = (const float*)d_in[8];
  const float* Wv = (const float*)d_in[9];  const float* bv = (const float*)d_in[10];
  const float* Wo = (const float*)d_in[11]; const float* bo = (const float*)d_in[12];

  // workspace layout (offsets multiple of 256B)
  char* ws = (char*)d_ws;
  __bf16* Qh = (__bf16*)(ws);                        // 7,372,800 B
  __bf16* Kh = (__bf16*)(ws + 7372800);              // 33,554,432 B
  __bf16* Vh = (__bf16*)(ws + 7372800 + 33554432);   // 16,777,216 B
  float* Oattn = (float*)(ws + 7372800 + 33554432 + 16777216);  // 7,372,800 B

  const int Mq = B_DIM * LQ;    // 7200
  const int Mk = B_DIM * LK;    // 32768

  dim3 g32(32);
  // Q = query @ Wq^T + bq  -> bf16 [B,H,Lq,64]
  gemm_wmma_bf16<<<dim3(Mq / 16, E_DIM / 64), g32, 0, stream>>>(
      query, Wq, bq, Mq, E_DIM, E_DIM, nullptr, Qh, LQ, HD);
  // K = key @ Wk^T + bk    -> bf16 [B,H,Lk,64]
  gemm_wmma_bf16<<<dim3(Mk / 16, E_DIM / 64), g32, 0, stream>>>(
      key, Wk, bk, Mk, E_DIM, E_DIM, nullptr, Kh, LK, HD);
  // V = value @ Wv^T + bv  -> bf16 [B,H,Lk,32]
  gemm_wmma_bf16<<<dim3(Mk / 16, V_DIM / 64), g32, 0, stream>>>(
      value, Wv, bv, Mk, V_DIM, V_DIM, nullptr, Vh, LK, VD);
  // flash attention -> fp32 [B,Lq,256]
  flash_attn<<<dim3((LQ + 63) / 64, B_DIM * H_DIM), dim3(128), 0, stream>>>(
      Qh, Kh, Vh, amask, kpm, Oattn);
  // out = Oattn @ Wo^T + bo -> fp32 d_out
  gemm_wmma_bf16<<<dim3(Mq / 16, V_DIM / 64), g32, 0, stream>>>(
      Oattn, Wo, bo, Mq, V_DIM, V_DIM, (float*)d_out, nullptr, 1, 1);
}
